// OT_Transport_83434034692457
// MI455X (gfx1250) — matrix operand order repaired and checked by reading, not compiled, not run on hardware
//
#include <hip/hip_runtime.h>
#include <hip/hip_bf16.h>
#include <math.h>

typedef __attribute__((ext_vector_type(16))) _Float16 v16h;
typedef __attribute__((ext_vector_type(8)))  float    v8f;

#define BM 256          // block tile M
#define BN 128          // block tile N
#define BK 32           // block tile K
#define THREADS 256     // 8 waves: 4 along M x 2 along N, wave tile 64x64

// ---------------- helpers ----------------

__device__ __forceinline__ unsigned pack2h(float a, float b) {
  union { _Float16 h[2]; unsigned u; } q;
  q.h[0] = (_Float16)a; q.h[1] = (_Float16)b;
  return q.u;
}

// Async 16-byte global -> LDS copy (no data VGPRs, tracked by ASYNCcnt).
// GV mode: vdst = LDS byte address (low 32 bits of generic pointer),
// vaddr = 64-bit global address, SADDR = off.
__device__ __forceinline__ void async_copy_b128(const _Float16* g, _Float16* l) {
  const unsigned           lds = (unsigned)(size_t)l;
  const unsigned long long ga  = (unsigned long long)(size_t)g;
  asm volatile("global_load_async_to_lds_b128 %0, %1, off"
               :: "v"(lds), "v"(ga) : "memory");
}

__device__ __forceinline__ void wait_async() {
  asm volatile("s_wait_asynccnt 0x0" ::: "memory");
}

// A fragment (16x32 f16, M x K). ISA: lanes 0-15: M=lane, K in {0..7,16..23};
// lanes 16-31: M=lane-16, K in {8..15,24..31}. rowbase -> ldsA[mrow*BK].
__device__ __forceinline__ v16h frag_a(const _Float16* rowbase, int hi) {
  const unsigned* p = (const unsigned*)rowbase;
  union { unsigned u[8]; v16h v; } f;
  const int h4 = hi * 4;
#pragma unroll
  for (int i = 0; i < 4; ++i) { f.u[i] = p[i + h4]; f.u[i + 4] = p[8 + i + h4]; }
  return f.v;
}

// B fragment (32x16 f16, K x N), LDS is N-major (ldsB[n*BK + k]).
// lanes 0-15: N=lane, K in {0..7,16..23}; lanes 16-31: N=lane-16, K in {8..15,24..31}.
__device__ __forceinline__ v16h frag_b(const _Float16* rowbase, int hi) {
  const unsigned* p = (const unsigned*)rowbase;
  union { unsigned u[8]; v16h v; } f;
  const int h4 = hi * 4;
#pragma unroll
  for (int i = 0; i < 4; ++i) { f.u[i] = p[i + h4]; f.u[i + 4] = p[8 + i + h4]; }
  return f.v;
}

// ---------------- one-shot fp32 -> fp16 conversion ----------------
__global__ __launch_bounds__(256)
void cvt_f16_kernel(const float* __restrict__ src, _Float16* __restrict__ dst) {
  const size_t t = (size_t)blockIdx.x * blockDim.x + threadIdx.x;
  const float4 a = ((const float4*)src)[2 * t];
  const float4 b = ((const float4*)src)[2 * t + 1];
  uint4 o;
  o.x = pack2h(a.x, a.y);
  o.y = pack2h(a.z, a.w);
  o.z = pack2h(b.x, b.y);
  o.w = pack2h(b.z, b.w);
  ((uint4*)dst)[t] = o;
}

// ---------------- one-shot fp32 [K,N] -> fp16 [N,K] transpose-convert --------
// 64x64 LDS tile, padded to kill bank conflicts; coalesced reads and writes.
__global__ __launch_bounds__(256)
void cvt_transpose_f16_kernel(const float* __restrict__ src, _Float16* __restrict__ dst,
                              int Kdim, int Ndim) {
  __shared__ _Float16 tile[64][65];
  const int bx = blockIdx.x * 64;   // N offset
  const int by = blockIdx.y * 64;   // K offset
  const int tid = threadIdx.x;
  const int rr = tid >> 4;          // 0..15
  const int cc = (tid & 15) << 2;   // 0..60
#pragma unroll
  for (int p = 0; p < 4; ++p) {
    const int r = rr + p * 16;      // k-local 0..63
    const float4 v = *(const float4*)(src + (size_t)(by + r) * Ndim + bx + cc);
    tile[cc + 0][r] = (_Float16)v.x;   // store transposed: tile[n_local][k_local]
    tile[cc + 1][r] = (_Float16)v.y;
    tile[cc + 2][r] = (_Float16)v.z;
    tile[cc + 3][r] = (_Float16)v.w;
  }
  __syncthreads();
#pragma unroll
  for (int p = 0; p < 4; ++p) {
    const int r = rr + p * 16;      // n-local 0..63
    union { _Float16 h[4]; uint2 u; } pk;
    pk.h[0] = tile[r][cc + 0];
    pk.h[1] = tile[r][cc + 1];
    pk.h[2] = tile[r][cc + 2];
    pk.h[3] = tile[r][cc + 3];
    *(uint2*)(dst + (size_t)(bx + r) * Kdim + by + cc) = pk.u;
  }
}

// ---------------- double-buffered WMMA GEMM (all-f16 operands) ----------------
// acc = A[M,K] @ B, epilogue per MODE:
//   MODE 0: + aux[N]                       (bias)
//   MODE 1: exp(acc * sm[m] * sn[n] * 20)  (cost -> Q)
//   MODE 2: + aux[m*N+n]                   (residual add)
// TRANSB=true:  B is [N,K] row-major (k-contiguous) -> fully async staging
// TRANSB=false: B is [K,N] row-major -> register loads + scatter transpose
// WF32: write fp32 C; WF16: write f16 Ch. Batched via blockIdx.z + strides.
template <int MODE, bool TRANSB, bool WF32, bool WF16>
__global__ __launch_bounds__(THREADS)
void gemm_wmma(const _Float16* __restrict__ A, const _Float16* __restrict__ Bm,
               float* __restrict__ C, _Float16* __restrict__ Ch,
               const float* __restrict__ aux,
               const float* __restrict__ scaleM, const float* __restrict__ scaleN,
               int M, int N, int K,
               long long sA, long long sB, long long sC, long long sCh,
               long long sAux, long long sSM, long long sSN) {
  __shared__ _Float16 ldsA[2 * BM * BK];   // [buf][m][k]  2 x 16KB
  __shared__ _Float16 ldsB[2 * BN * BK];   // [buf][n][k]  2 x 8KB

  const int z = blockIdx.z;
  const _Float16* Ab = A + (size_t)z * sA;
  const _Float16* Bb = Bm + (size_t)z * sB;
  float*     Cb  = WF32 ? (C + (size_t)z * sC) : nullptr;
  _Float16*  Chb = WF16 ? (Ch + (size_t)z * sCh) : nullptr;
  const float* auxb = (MODE != 1) ? (aux + (size_t)z * sAux) : nullptr;
  const float* smb  = (MODE == 1) ? (scaleM + (size_t)z * sSM) : nullptr;
  const float* snb  = (MODE == 1) ? (scaleN + (size_t)z * sSN) : nullptr;

  const int tid   = threadIdx.x;
  const int lane  = tid & 31;
  const int wave  = tid >> 5;
  const int wm    = wave & 3;   // 4 waves along M (64 rows each)
  const int wn    = wave >> 2;  // 2 waves along N (64 cols each)
  const int lhalf = lane & 15;
  const int hi    = lane >> 4;

  const int m0 = blockIdx.y * BM;
  const int n0 = blockIdx.x * BN;

  // ---- tile movers ---------------------------------------------------------
  // A tile: 256x32 halfs = 1024 x 16B chunks; 4 async copies per thread.
  auto async_A = [&](int kb, int buf) {
    _Float16* lA = ldsA + buf * (BM * BK);
#pragma unroll
    for (int it = 0; it < 4; ++it) {
      int idx = tid + it * THREADS;          // 0..1023
      int m   = idx >> 2;                    // 0..255
      int ko  = (idx & 3) << 3;              // 0,8,16,24
      async_copy_b128(Ab + (size_t)(m0 + m) * K + kb + ko, &lA[m * BK + ko]);
    }
  };
  // TRANSB B tile ([N,K], k-contiguous): 128x32 halfs; 2 async copies/thread.
  auto async_B = [&](int kb, int buf) {
    _Float16* lB = ldsB + buf * (BN * BK);
#pragma unroll
    for (int it = 0; it < 2; ++it) {
      int idx = tid + it * THREADS;          // 0..511
      int n   = idx >> 2;                    // 0..127
      int ko  = (idx & 3) << 3;
      async_copy_b128(Bb + (size_t)(n0 + n) * K + kb + ko, &lB[n * BK + ko]);
    }
  };
  // [K,N] B tile: coalesced 8-half register loads, scatter-transpose to LDS.
  uint4 rb[2];
  auto load_B = [&](int kb) {
#pragma unroll
    for (int it = 0; it < 2; ++it) {
      int idx = tid + it * THREADS;          // 0..511
      int k   = idx >> 4;                    // 0..31
      int no  = (idx & 15) << 3;             // 0..120 step 8
      rb[it] = *(const uint4*)(Bb + (size_t)(kb + k) * N + n0 + no);
    }
  };
  auto store_B = [&](int buf) {
    _Float16* lB = ldsB + buf * (BN * BK);
#pragma unroll
    for (int it = 0; it < 2; ++it) {
      int idx = tid + it * THREADS;
      int k   = idx >> 4;
      int no  = (idx & 15) << 3;
      union { uint4 q; _Float16 h[8]; } u;
      u.q = rb[it];
#pragma unroll
      for (int j = 0; j < 8; ++j)
        lB[(no + j) * BK + k] = u.h[j];
    }
  };

  // ---- accumulators --------------------------------------------------------
  v8f acc[4][4];
#pragma unroll
  for (int mi = 0; mi < 4; ++mi)
#pragma unroll
    for (int ni = 0; ni < 4; ++ni)
      acc[mi][ni] = (v8f){0.f, 0.f, 0.f, 0.f, 0.f, 0.f, 0.f, 0.f};

  // ---- prologue: stage tile 0 ----------------------------------------------
  async_A(0, 0);
  if (TRANSB) {
    async_B(0, 0);
  } else {
    load_B(0);
    store_B(0);
  }
  wait_async();
  __syncthreads();

  // ---- main loop: compute buf while async copies fill buf^1 ----------------
  int buf = 0;
  for (int kb = 0; kb < K; kb += BK, buf ^= 1) {
    const bool has_next = (kb + BK) < K;
    if (has_next) {
      async_A(kb + BK, buf ^ 1);             // fire-and-forget into other buffer
      if (TRANSB) async_B(kb + BK, buf ^ 1);
      else        load_B(kb + BK);           // issue register loads, no wait
    }

    const _Float16* lA = ldsA + buf * (BM * BK);
    const _Float16* lB = ldsB + buf * (BN * BK);

    v16h bf[4];
#pragma unroll
    for (int ni = 0; ni < 4; ++ni)
      bf[ni] = frag_b(&lB[(wn * 64 + ni * 16 + lhalf) * BK], hi);
#pragma unroll
    for (int mi = 0; mi < 4; ++mi) {
      const v16h af = frag_a(&lA[(wm * 64 + mi * 16 + lhalf) * BK], hi);
#pragma unroll
      for (int ni = 0; ni < 4; ++ni)
        acc[mi][ni] = __builtin_amdgcn_wmma_f32_16x16x32_f16(
            false, af, false, bf[ni], (short)0, acc[mi][ni], false, false);
    }

    if (has_next) {
      if (!TRANSB) store_B(buf ^ 1);         // B-load latency covered by WMMAs
      wait_async();                          // drain this wave's async copies
    }
    __syncthreads();                         // one barrier per K-step
  }

  // ---- epilogue: C 16x16 layout: VGPR r -> row r + hi*8, col = lhalf -------
#pragma unroll
  for (int mi = 0; mi < 4; ++mi) {
#pragma unroll
    for (int ni = 0; ni < 4; ++ni) {
      const int grow0 = m0 + wm * 64 + mi * 16 + hi * 8;
      const int gcol  = n0 + wn * 64 + ni * 16 + lhalf;
#pragma unroll
      for (int r = 0; r < 8; ++r) {
        const int grow = grow0 + r;
        float v = acc[mi][ni][r];
        if (MODE == 0)      v += auxb[gcol];
        else if (MODE == 1) v = __expf(v * smb[grow] * snb[gcol] * 20.0f); // /OT_REG
        else                v += auxb[(size_t)grow * N + gcol];
        const size_t oidx = (size_t)grow * N + gcol;
        if (WF32) Cb[oidx]  = v;
        if (WF16) Chb[oidx] = (_Float16)v;
      }
    }
  }
}

// ---------------- small kernels ----------------

// out[row] = 1 / max(||X[row,:]||_2, eps)   (X is f16)
__global__ __launch_bounds__(256)
void rownorm_inv_kernel(const _Float16* __restrict__ X, float* __restrict__ out,
                        int rowlen, float eps) {
  const _Float16* row = X + (size_t)blockIdx.x * rowlen;
  const int tid = threadIdx.x;
  float s = 0.f;
  for (int i = tid; i < rowlen; i += 256) { float v = (float)row[i]; s += v * v; }
  __shared__ float red[256];
  red[tid] = s; __syncthreads();
  for (int off = 128; off > 0; off >>= 1) {
    if (tid < off) red[tid] += red[tid + off];
    __syncthreads();
  }
  if (tid == 0) out[blockIdx.x] = 1.f / fmaxf(sqrtf(red[0]), eps);
}

// row /= sum(row)  (Sinkhorn row step, values positive)
__global__ __launch_bounds__(256)
void row_div_kernel(float* __restrict__ Q, int rowlen) {
  float* row = Q + (size_t)blockIdx.x * rowlen;
  const int tid = threadIdx.x;
  float s = 0.f;
  for (int i = tid; i < rowlen; i += 256) s += row[i];
  __shared__ float red[256];
  red[tid] = s; __syncthreads();
  for (int off = 128; off > 0; off >>= 1) {
    if (tid < off) red[tid] += red[tid + off];
    __syncthreads();
  }
  const float inv = 1.f / red[0];
  for (int i = tid; i < rowlen; i += 256) row[i] *= inv;
}

// Q[b, :, o] /= sum_t Q[b, t, o]  (thread <-> column, fully coalesced)
__global__ __launch_bounds__(256)
void col_div_kernel(float* __restrict__ Q, int T, int O) {
  const int o = blockIdx.x * blockDim.x + threadIdx.x;
  float* base = Q + (size_t)blockIdx.y * T * O;
  float s = 0.f;
  for (int t = 0; t < T; ++t) s += base[(size_t)t * O + o];
  const float inv = 1.f / s;
  for (int t = 0; t < T; ++t) base[(size_t)t * O + o] *= inv;
}

// plan = l2norm(Q, dim=-1) emitted directly as f16 (feeds the final GEMM's A)
__global__ __launch_bounds__(256)
void plan_f16_kernel(const float* __restrict__ Q, _Float16* __restrict__ P, int rowlen) {
  const float* row = Q + (size_t)blockIdx.x * rowlen;
  _Float16* prow = P + (size_t)blockIdx.x * rowlen;
  const int tid = threadIdx.x;
  float s = 0.f;
  for (int i = tid; i < rowlen; i += 256) { float v = row[i]; s += v * v; }
  __shared__ float red[256];
  red[tid] = s; __syncthreads();
  for (int off = 128; off > 0; off >>= 1) {
    if (tid < off) red[tid] += red[tid + off];
    __syncthreads();
  }
  const float inv = 1.f / fmaxf(sqrtf(red[0]), 1e-12f);
  for (int i = tid; i < rowlen; i += 256) prow[i] = (_Float16)(row[i] * inv);
}

// LayerNorm over last dim H with gamma/beta
__global__ __launch_bounds__(256)
void layernorm_kernel(const float* __restrict__ X, const float* __restrict__ g,
                      const float* __restrict__ b, float* __restrict__ out, int H) {
  const float* row = X + (size_t)blockIdx.x * H;
  float* orow = out + (size_t)blockIdx.x * H;
  const int tid = threadIdx.x;
  float s1 = 0.f, s2 = 0.f;
  for (int i = tid; i < H; i += 256) {
    float v = row[i];
    s1 += v; s2 += v * v;
  }
  __shared__ float r1[256], r2[256];
  r1[tid] = s1; r2[tid] = s2; __syncthreads();
  for (int off = 128; off > 0; off >>= 1) {
    if (tid < off) { r1[tid] += r1[tid + off]; r2[tid] += r2[tid + off]; }
    __syncthreads();
  }
  const float mu  = r1[0] / (float)H;
  const float var = r2[0] / (float)H - mu * mu;
  const float rstd = rsqrtf(var + 1e-5f);
  for (int i = tid; i < H; i += 256)
    orow[i] = (row[i] - mu) * rstd * g[i] + b[i];
}

// ---------------- host orchestration ----------------

extern "C" void kernel_launch(void* const* d_in, const int* in_sizes, int n_in,
                              void* d_out, int out_size, void* d_ws, size_t ws_size,
                              hipStream_t stream) {
  (void)in_sizes; (void)n_in; (void)out_size; (void)ws_size;

  constexpr int NB = 16, L = 512, D = 1024;
  constexpr int ROWS = NB * L;                       // 8192
  constexpr long long MAT = (long long)ROWS * D;     // 8,388,608
  constexpr long long WSZ = (long long)D * D;        // 1,048,576
  constexpr long long QSZ = (long long)NB * L * L;   // 4,194,304

  const float* Xo    = (const float*)d_in[0];
  const float* Xt    = (const float*)d_in[1];
  const float* W_ori = (const float*)d_in[2];
  const float* b_ori = (const float*)d_in[3];
  const float* W_tar = (const float*)d_in[4];
  const float* b_tar = (const float*)d_in[5];
  const float* W_ft  = (const float*)d_in[6];
  const float* b_ft  = (const float*)d_in[7];
  const float* W_fo  = (const float*)d_in[8];
  const float* b_fo  = (const float*)d_in[9];
  const float* W_fot = (const float*)d_in[10];
  const float* b_fot = (const float*)d_in[11];
  const float* ln_g  = (const float*)d_in[12];
  const float* ln_b  = (const float*)d_in[13];

  // ---- bump allocator over workspace (256B aligned blocks)
  char* cur = (char*)d_ws;
  auto alloc = [&](size_t bytes) -> void* {
    void* r = (void*)cur;
    cur += (bytes + 255) & ~(size_t)255;
    return r;
  };

  float*     tar      = (float*)alloc(MAT * 4);       // fp32 tar (residual + LN)
  float*     Q        = (float*)alloc(QSZ * 4);       // Sinkhorn matrix
  _Float16*  Xo_h     = (_Float16*)alloc(MAT * 2);    // } contiguous 32MB, reused as
  _Float16*  Xt_h     = (_Float16*)alloc(MAT * 2);    // } fp32 out_pre at the end
  _Float16*  Wt_ori   = (_Float16*)alloc(WSZ * 2);    // transposed f16 weights [N,K]
  _Float16*  Wt_tar   = (_Float16*)alloc(WSZ * 2);
  _Float16*  Wt_ft    = (_Float16*)alloc(WSZ * 2);
  _Float16*  Wt_fo    = (_Float16*)alloc(WSZ * 2);
  _Float16*  Wt_fot   = (_Float16*)alloc(WSZ * 2);
  _Float16*  ori_h    = (_Float16*)alloc(MAT * 2);
  _Float16*  bufB_h   = (_Float16*)alloc(MAT * 2);    // tar1_h, then orifc_h
  _Float16*  tar_h    = (_Float16*)alloc(MAT * 2);
  _Float16*  oritp_h  = (_Float16*)alloc(MAT * 2);
  _Float16*  plan_h   = (_Float16*)alloc(QSZ * 2);
  float*     inv_nt   = (float*)alloc(ROWS * 4);
  float*     inv_no   = (float*)alloc(ROWS * 4);
  float*     outpre   = (float*)Xo_h;                 // alias: Xo_h/Xt_h dead by then

  const dim3 blk(THREADS);
  const dim3 gProj(D / BN, ROWS / BM, 1);             // (8, 32, 1)
  const dim3 gT(D / 64, D / 64, 1);                   // weight transpose grid

  // ---- one-shot conversions (activations straight, weights transposed)
  cvt_f16_kernel<<<(int)(MAT / 2048), 256, 0, stream>>>(Xo, Xo_h);
  cvt_f16_kernel<<<(int)(MAT / 2048), 256, 0, stream>>>(Xt, Xt_h);
  cvt_transpose_f16_kernel<<<gT, 256, 0, stream>>>(W_ori, Wt_ori, D, D);
  cvt_transpose_f16_kernel<<<gT, 256, 0, stream>>>(W_tar, Wt_tar, D, D);
  cvt_transpose_f16_kernel<<<gT, 256, 0, stream>>>(W_ft,  Wt_ft,  D, D);
  cvt_transpose_f16_kernel<<<gT, 256, 0, stream>>>(W_fo,  Wt_fo,  D, D);
  cvt_transpose_f16_kernel<<<gT, 256, 0, stream>>>(W_fot, Wt_fot, D, D);

  // ---- projection GEMMs: fully-async staging (TRANSB=true, weights [N,K])
  gemm_wmma<0, true, false, true><<<gProj, blk, 0, stream>>>(
      Xo_h, Wt_ori, nullptr, ori_h, b_ori, nullptr, nullptr,
      ROWS, D, D, 0, 0, 0, 0, 0, 0, 0);
  gemm_wmma<0, true, false, true><<<gProj, blk, 0, stream>>>(
      Xt_h, Wt_tar, nullptr, bufB_h, b_tar, nullptr, nullptr,
      ROWS, D, D, 0, 0, 0, 0, 0, 0, 0);
  gemm_wmma<0, true, true, true><<<gProj, blk, 0, stream>>>(
      bufB_h, Wt_ft, tar, tar_h, b_ft, nullptr, nullptr,
      ROWS, D, D, 0, 0, 0, 0, 0, 0, 0);
  gemm_wmma<0, true, false, true><<<gProj, blk, 0, stream>>>(
      ori_h, Wt_fo, nullptr, bufB_h, b_fo, nullptr, nullptr,   // bufB_h = ori_fc
      ROWS, D, D, 0, 0, 0, 0, 0, 0, 0);
  gemm_wmma<0, true, false, true><<<gProj, blk, 0, stream>>>(
      ori_h, Wt_fot, nullptr, oritp_h, b_fot, nullptr, nullptr,
      ROWS, D, D, 0, 0, 0, 0, 0, 0, 0);

  // ---- row inverse norms for cosine similarity
  rownorm_inv_kernel<<<ROWS, 256, 0, stream>>>(tar_h,  inv_nt, D, 1e-8f);
  rownorm_inv_kernel<<<ROWS, 256, 0, stream>>>(bufB_h, inv_no, D, 1e-8f);

  // ---- cost GEMM + exp(C/0.05) -> Q (batched over 16, fully async)
  gemm_wmma<1, true, true, false><<<dim3(L / BN, L / BM, NB), blk, 0, stream>>>(
      tar_h, bufB_h, Q, nullptr, nullptr, inv_nt, inv_no,
      L, L, D,
      (long long)L * D, (long long)L * D, (long long)L * L, 0, 0, L, L);

  // ---- Sinkhorn: 5 x (row normalize, col normalize)
  for (int it = 0; it < 5; ++it) {
    row_div_kernel<<<NB * L, 256, 0, stream>>>(Q, L);
    col_div_kernel<<<dim3(L / 256, NB), 256, 0, stream>>>(Q, L, L);
  }
  // ---- plan = l2norm(Q, dim=-1), emitted as f16
  plan_f16_kernel<<<NB * L, 256, 0, stream>>>(Q, plan_h, L);

  // ---- out_pre = plan @ ori_tp + tar (batched; B is [K,N] -> scatter path)
  gemm_wmma<2, false, true, false><<<dim3(D / BN, L / BM, NB), blk, 0, stream>>>(
      plan_h, oritp_h, outpre, nullptr, tar, nullptr, nullptr,
      L, D, L,
      (long long)L * L, (long long)L * D, (long long)L * D, 0, (long long)L * D, 0, 0);

  // ---- LayerNorm -> d_out
  layernorm_kernel<<<NB * L, 256, 0, stream>>>(outpre, ln_g, ln_b, (float*)d_out, D);
}
